// DeforAttn_10892037062988
// MI455X (gfx1250) — matrix operand order repaired and checked by reading, not compiled, not run on hardware
//
#include <hip/hip_runtime.h>
#include <stdint.h>

// ---------------- problem constants ----------------
#define NQ   16384
#define NV   21760
#define CDIM 256
#define NH   8
#define NPNT 4
#define NLVL 4

typedef _Float16 f16_t;
typedef __attribute__((ext_vector_type(16))) _Float16 v16h;
typedef __attribute__((ext_vector_type(8)))  float    v8f;

// =====================================================================
// Pack an MxK (K=256) row-major f32 matrix (optionally sum of two f32
// matrices) into the wave32 WMMA A-operand layout, split into f16 hi/lo.
// A layout (16x32 f16 tile): lane = hk*16 + m (m=0..15, hk=0..1),
//   element e (0..15): K = kchunk*32 + 8*hk + e + ((e>>3)<<3)
// Packed storage: vec16h index = (mtile*8 + kchunk)*32 + lane
// =====================================================================
__global__ __launch_bounds__(256) void pack_a_kernel(
    const float* __restrict__ src0, const float* __restrict__ src1,
    f16_t* __restrict__ hi, f16_t* __restrict__ lo, int mtiles) {
  int wave = (blockIdx.x * blockDim.x + threadIdx.x) >> 5;
  int lane = threadIdx.x & 31;
  int mt = wave >> 3;
  int kc = wave & 7;
  if (mt >= mtiles) return;
  int m  = mt * 16 + (lane & 15);
  int hk = lane >> 4;
  const float* r0 = src0 + (size_t)m * CDIM;
  const float* r1 = src1 ? (src1 + (size_t)m * CDIM) : nullptr;
  v16h vhi, vlo;
#pragma unroll
  for (int e = 0; e < 16; ++e) {
    int k = kc * 32 + 8 * hk + e + ((e >> 3) << 3);
    float x = r0[k];
    if (r1) x += r1[k];
    f16_t h = (f16_t)x;
    vhi[e] = h;
    vlo[e] = (f16_t)(x - (float)h);
  }
  size_t o = ((size_t)(mt * 8 + kc) * 32 + lane) * 16;
  *reinterpret_cast<v16h*>(hi + o) = vhi;
  *reinterpret_cast<v16h*>(lo + o) = vlo;
}

// =====================================================================
// Pack a KxN (K=256) row-major f32 weight into wave32 WMMA B-operand
// layout (32x16 f16 tile): lane = hk*16 + n, element e: K = 8*hk + e + ((e>>3)<<3).
// Packed storage: vec16h index = (ntile*8 + kchunk)*32 + lane
// =====================================================================
__global__ __launch_bounds__(256) void pack_b_kernel(
    const float* __restrict__ W, int N,
    f16_t* __restrict__ hi, f16_t* __restrict__ lo, int ntiles) {
  int wave = (blockIdx.x * blockDim.x + threadIdx.x) >> 5;
  int lane = threadIdx.x & 31;
  int nt = wave >> 3;
  int kc = wave & 7;
  if (nt >= ntiles) return;
  int n  = nt * 16 + (lane & 15);
  int hk = lane >> 4;
  v16h vhi, vlo;
#pragma unroll
  for (int e = 0; e < 16; ++e) {
    int k = kc * 32 + 8 * hk + e + ((e >> 3) << 3);
    float x = W[(size_t)k * N + n];
    f16_t h = (f16_t)x;
    vhi[e] = h;
    vlo[e] = (f16_t)(x - (float)h);
  }
  size_t o = ((size_t)(nt * 8 + kc) * 32 + lane) * 16;
  *reinterpret_cast<v16h*>(hi + o) = vhi;
  *reinterpret_cast<v16h*>(lo + o) = vlo;
}

// =====================================================================
// GEMM: C[M,N] = A[M,256] * B[256,N] + bias (+ residual), fp32-accurate
// via f16 hi/lo split (3x v_wmma_f32_16x16x32_f16 per K-chunk).
// One wave -> 16x64 output tile (4 accumulators), K loop of 8 chunks.
// =====================================================================
__global__ __launch_bounds__(256) void gemm_hilo_kernel(
    const f16_t* __restrict__ Ahi, const f16_t* __restrict__ Alo,
    const f16_t* __restrict__ Bhi, const f16_t* __restrict__ Blo,
    const float* __restrict__ bias, const float* __restrict__ residual,
    float* __restrict__ C, int mtiles, int nquads, int ldc) {
  int wave = (blockIdx.x * blockDim.x + threadIdx.x) >> 5;
  int lane = threadIdx.x & 31;
  int mt = wave / nquads;
  int nq = wave % nquads;
  if (mt >= mtiles) return;

  const v16h* pAh = reinterpret_cast<const v16h*>(Ahi);
  const v16h* pAl = reinterpret_cast<const v16h*>(Alo);
  const v16h* pBh = reinterpret_cast<const v16h*>(Bhi);
  const v16h* pBl = reinterpret_cast<const v16h*>(Blo);

  v8f acc[4];
#pragma unroll
  for (int j = 0; j < 4; ++j) acc[j] = (v8f)(0.0f);

#pragma unroll 2
  for (int kc = 0; kc < 8; ++kc) {
    size_t ai = ((size_t)mt * 8 + kc) * 32 + lane;
    v16h ah = pAh[ai];
    v16h al = pAl[ai];
#pragma unroll
    for (int j = 0; j < 4; ++j) {
      int nt = nq * 4 + j;
      size_t bi = ((size_t)nt * 8 + kc) * 32 + lane;
      v16h bh = pBh[bi];
      v16h bl = pBl[bi];
      // hi*hi + lo*hi + hi*lo  (lo*lo dropped: ~2^-22 relative)
      acc[j] = __builtin_amdgcn_wmma_f32_16x16x32_f16(false, ah, false, bh,
                                                      (short)0, acc[j], false, false);
      acc[j] = __builtin_amdgcn_wmma_f32_16x16x32_f16(false, al, false, bh,
                                                      (short)0, acc[j], false, false);
      acc[j] = __builtin_amdgcn_wmma_f32_16x16x32_f16(false, ah, false, bl,
                                                      (short)0, acc[j], false, false);
    }
  }

  // C/D layout: VGPR r -> row = mt*16 + 8*(lane>>4) + r, col = lane&15
  int hk = lane >> 4;
  int nl = lane & 15;
#pragma unroll
  for (int j = 0; j < 4; ++j) {
    int col = (nq * 4 + j) * 16 + nl;
    float bv = bias ? bias[col] : 0.0f;
#pragma unroll
    for (int r = 0; r < 8; ++r) {
      int row = mt * 16 + hk * 8 + r;
      float v = acc[j][r] + bv;
      if (residual) v += residual[(size_t)row * ldc + col];
      C[(size_t)row * ldc + col] = v;
    }
  }
}

// =====================================================================
// Deformable sampling core. One block per query; wave w = head w;
// lane = channel (d=32 == wave32). Softmax(16) via __shfl_xor.
// Prologue uses CDNA5 async global->LDS copies (ASYNCcnt) to fill the
// per-query offsets/logits/refpoints directly into LDS (no VGPR staging):
// lanes 0..95 cover the 1536B offattn row in 16B chunks, lanes 96..97
// cover the 32B refpoint row; then s_wait_asynccnt 0 + workgroup barrier.
// v_proj (22MB) is L2-resident; each corner read is a coalesced
// 128B wave access: v_proj[(start+idx)*256 + head*32 + lane].
// offattn row: [0..255] = offsets (h,l,p,2), [256..383] = attn logits.
// =====================================================================
__global__ __launch_bounds__(256) void msda_sample_kernel(
    const float* __restrict__ offattn, const float* __restrict__ vproj,
    const float* __restrict__ refpts, float* __restrict__ sampled) {
  __shared__ float s_buf[392];  // [0,256)=off  [256,384)=attn  [384,392)=ref
  float* s_off  = s_buf;
  float* s_attn = s_buf + 256;
  float* s_ref  = s_buf + 384;

  int q   = blockIdx.x;
  int tid = threadIdx.x;

  // ---- async global -> LDS fill (tracked by ASYNCcnt) ----
  unsigned lds_base = (unsigned)(uintptr_t)(&s_buf[0]);  // LDS addr = flat[31:0]
  if (tid < 96) {
    unsigned long long gp =
        (unsigned long long)(uintptr_t)((const char*)(offattn + (size_t)q * 384) + tid * 16);
    unsigned lds_addr = lds_base + (unsigned)tid * 16u;
    asm volatile("global_load_async_to_lds_b128 %0, %1, off"
                 :: "v"(lds_addr), "v"(gp) : "memory");
  } else if (tid < 98) {
    unsigned long long gp =
        (unsigned long long)(uintptr_t)((const char*)(refpts + (size_t)q * 8) + (tid - 96) * 16);
    unsigned lds_addr = lds_base + 384u * 4u + (unsigned)(tid - 96) * 16u;
    asm volatile("global_load_async_to_lds_b128 %0, %1, off"
                 :: "v"(lds_addr), "v"(gp) : "memory");
  }
  asm volatile("s_wait_asynccnt 0x0" ::: "memory");
  __syncthreads();

  int h    = tid >> 5;
  int lane = tid & 31;

  // per-head softmax over 16 logits (lanes 16-31 mirror lanes 0-15)
  float lg = s_attn[h * 16 + (lane & 15)];
  float mx = lg;
#pragma unroll
  for (int s = 8; s > 0; s >>= 1) mx = fmaxf(mx, __shfl_xor(mx, s, 32));
  float ex = __expf(lg - mx);
  float sm = ex;
#pragma unroll
  for (int s = 8; s > 0; s >>= 1) sm += __shfl_xor(sm, s, 32);
  float wsm = ex / sm;
  if (lane < 16) s_attn[h * 16 + lane] = wsm;
  __syncthreads();

  const int HS[4] = {128, 64, 32, 16};
  const int ST[4] = {0, 16384, 20480, 21504};

  float acc = 0.0f;
#pragma unroll
  for (int lvl = 0; lvl < NLVL; ++lvl) {
    const int   Wi = HS[lvl], Hi = HS[lvl];          // square levels
    const float Wf = (float)Wi, Hf = (float)Hi;
    float rx = s_ref[lvl * 2 + 0];
    float ry = s_ref[lvl * 2 + 1];
#pragma unroll
    for (int p = 0; p < NPNT; ++p) {
      float ox = s_off[((h * NLVL + lvl) * NPNT + p) * 2 + 0];
      float oy = s_off[((h * NLVL + lvl) * NPNT + p) * 2 + 1];
      float aw = s_attn[h * 16 + lvl * NPNT + p];
      float x  = rx * Wf + ox - 0.5f;                 // (rx + ox/W)*W - 0.5
      float y  = ry * Hf + oy - 0.5f;
      float x0 = floorf(x), y0 = floorf(y);
      float lx = x - x0,    ly = y - y0;
      int xi = (int)x0, yi = (int)y0;
#pragma unroll
      for (int c = 0; c < 4; ++c) {
        int dx = c & 1, dy = c >> 1;
        int cx = xi + dx, cy = yi + dy;
        bool valid = (cx >= 0) & (cx < Wi) & (cy >= 0) & (cy < Hi);
        int ccx = cx < 0 ? 0 : (cx > Wi - 1 ? Wi - 1 : cx);
        int ccy = cy < 0 ? 0 : (cy > Hi - 1 ? Hi - 1 : cy);
        float bw = (dx ? lx : 1.0f - lx) * (dy ? ly : 1.0f - ly);
        float vv = vproj[((size_t)(ST[lvl] + ccy * Wi + ccx)) * CDIM + h * 32 + lane];
        acc += aw * bw * (valid ? vv : 0.0f);
      }
    }
  }
  sampled[(size_t)q * CDIM + tid] = acc;
}

// =====================================================================
// Host-side orchestration (graph-capture safe: only kernel launches and
// hipMemcpyAsync d2d on `stream`).
// =====================================================================
extern "C" void kernel_launch(void* const* d_in, const int* in_sizes, int n_in,
                              void* d_out, int out_size, void* d_ws, size_t ws_size,
                              hipStream_t stream) {
  const float* query  = (const float*)d_in[0];
  const float* qpos   = (const float*)d_in[1];
  const float* value  = (const float*)d_in[2];
  const float* refp   = (const float*)d_in[3];
  const float* W_off  = (const float*)d_in[4];
  const float* b_off  = (const float*)d_in[5];
  const float* W_attn = (const float*)d_in[6];
  const float* b_attn = (const float*)d_in[7];
  const float* W_val  = (const float*)d_in[8];
  const float* b_val  = (const float*)d_in[9];
  const float* W_out  = (const float*)d_in[10];
  const float* b_out  = (const float*)d_in[11];
  float* out = (float*)d_out;

  // ---- workspace carve-up (256B aligned) ----
  char* ws = (char*)d_ws;
  size_t off = 0;
  auto carve = [&](size_t bytes) {
    char* p = ws + off;
    off += (bytes + 255) & ~(size_t)255;
    return p;
  };
  f16_t* qA_hi   = (f16_t*)carve((size_t)NQ * CDIM * 2);   // also reused as sampled-A hi
  f16_t* qA_lo   = (f16_t*)carve((size_t)NQ * CDIM * 2);   // also reused as sampled-A lo
  f16_t* vA_hi   = (f16_t*)carve((size_t)NV * CDIM * 2);   // region reused for `sampled` f32
  f16_t* vA_lo   = (f16_t*)carve((size_t)NV * CDIM * 2);
  f16_t* Bval_hi = (f16_t*)carve((size_t)CDIM * CDIM * 2);
  f16_t* Bval_lo = (f16_t*)carve((size_t)CDIM * CDIM * 2);
  f16_t* Boa_hi  = (f16_t*)carve((size_t)CDIM * 384 * 2);
  f16_t* Boa_lo  = (f16_t*)carve((size_t)CDIM * 384 * 2);
  f16_t* Bout_hi = (f16_t*)carve((size_t)CDIM * CDIM * 2);
  f16_t* Bout_lo = (f16_t*)carve((size_t)CDIM * CDIM * 2);
  float* bias_oa = (float*)carve(384 * 4);
  float* v_proj  = (float*)carve((size_t)NV * CDIM * 4);
  float* offattn = (float*)carve((size_t)NQ * 384 * 4);
  // sampled f32 aliases the (dead-after-GEMM1) value-A pack region
  float* sampled = (float*)vA_hi;                          // NQ*256*4 <= 2*NV*256*2

  const int TPB = 256;

  // 1) pack A operands: q = query + query_pos ; value
  pack_a_kernel<<<(NQ / 16) * 8 / 8, TPB, 0, stream>>>(query, qpos, qA_hi, qA_lo, NQ / 16);
  pack_a_kernel<<<(NV / 16) * 8 / 8, TPB, 0, stream>>>(value, nullptr, vA_hi, vA_lo, NV / 16);

  // 2) pack B operands (W_off and W_attn fused into one N=384 operand)
  pack_b_kernel<<<16, TPB, 0, stream>>>(W_val, CDIM, Bval_hi, Bval_lo, 16);
  pack_b_kernel<<<16, TPB, 0, stream>>>(W_off, CDIM, Boa_hi, Boa_lo, 16);
  // W_attn tiles land at ntile offset 16 -> pointer offset 16*8*32*16 halves
  pack_b_kernel<<<8, TPB, 0, stream>>>(W_attn, 128,
                                       Boa_hi + (size_t)16 * 8 * 32 * 16,
                                       Boa_lo + (size_t)16 * 8 * 32 * 16, 8);
  pack_b_kernel<<<16, TPB, 0, stream>>>(W_out, CDIM, Bout_hi, Bout_lo, 16);

  // 3) concat bias [b_off | b_attn]
  hipMemcpyAsync(bias_oa,       b_off,  256 * sizeof(float), hipMemcpyDeviceToDevice, stream);
  hipMemcpyAsync(bias_oa + 256, b_attn, 128 * sizeof(float), hipMemcpyDeviceToDevice, stream);

  // 4) GEMM1: v_proj[NV,256] = value @ W_val + b_val
  gemm_hilo_kernel<<<(NV / 16) * 4 / 8, TPB, 0, stream>>>(
      vA_hi, vA_lo, Bval_hi, Bval_lo, b_val, nullptr, v_proj, NV / 16, 4, CDIM);

  // 5) GEMM2: offattn[NQ,384] = q @ [W_off|W_attn] + [b_off|b_attn]
  gemm_hilo_kernel<<<(NQ / 16) * 6 / 8, TPB, 0, stream>>>(
      qA_hi, qA_lo, Boa_hi, Boa_lo, bias_oa, nullptr, offattn, NQ / 16, 6, 384);

  // 6) deformable bilinear sampling -> sampled[NQ,256]
  msda_sample_kernel<<<NQ, TPB, 0, stream>>>(offattn, v_proj, refp, sampled);

  // 7) repack sampled as A operand (reuses qA buffers, dead after GEMM2)
  pack_a_kernel<<<(NQ / 16) * 8 / 8, TPB, 0, stream>>>(sampled, nullptr, qA_hi, qA_lo, NQ / 16);

  // 8) GEMM3: out = sampled @ W_out + b_out + query (identity residual)
  gemm_hilo_kernel<<<(NQ / 16) * 4 / 8, TPB, 0, stream>>>(
      qA_hi, qA_lo, Bout_hi, Bout_lo, b_out, query, out, NQ / 16, 4, CDIM);
}
